// STGCNBlock_9646496547016
// MI455X (gfx1250) — matrix operand order repaired and checked by reading, not compiled, not run on hardware
//
#include <hip/hip_runtime.h>
#include <hip/hip_bf16.h>

// ---------------------------------------------------------------------------
// STGCN block for MI455X (gfx1250, wave32, WMMA).
// B=4, Cin=32, N=1024, S=32, Cout=64, Csp=64, K=3, E=16384
// All GEMM-shaped math on v_wmma_f32_16x16x32_bf16 (bf16 A/B, f32 acc).
// LDS tiles are stored fragment-major so each lane's 16-bf16 fragment is one
// contiguous 32B v16bf load (2x ds_load_b128), not 16 scalar ds_load_u16.
// ---------------------------------------------------------------------------

#define BB   4
#define CIN  32
#define NN   1024
#define SS   32
#define COUT 64
#define CSP  64
#define KW   3
#define EPSV 1e-5f

typedef __attribute__((ext_vector_type(16))) __bf16 v16bf;
typedef __attribute__((ext_vector_type(8)))  float  v8f;

// ------------------- fragment-major LDS swizzle maps -----------------------
// A-matrix fragment map (16x32 bf16 per tile):  K(e) = (e<8?0:16)+hi*8+(e&7)
// inverse: hi=(k>>3)&1, e=(k&7)|((k>>4)<<3), lane=hi*16+(m&15)
// storage: [mtile][kstep][lane(32)][e(16)]
__device__ __forceinline__ int a_swz(int m, int k, int ksteps) {
  const int mtile = m >> 4, ml = m & 15;
  const int kstep = k >> 5, kk = k & 31;
  const int hi = (kk >> 3) & 1;
  const int e  = (kk & 7) | ((kk >> 4) << 3);
  return (((mtile * ksteps) + kstep) * 32 + (hi * 16 + ml)) * 16 + e;
}

// B-matrix fragment map (32x16 bf16 per tile):  K(e) = (e<8?0:16)+hi*8+(e&7)
// inverse: hi=(k>>3)&1, e=(k&7)|((k>>4)<<3), lane=hi*16+(n&15)
// storage: [kstep][ntile][lane(32)][e(16)]
__device__ __forceinline__ int b_swz(int k, int n, int ntiles) {
  const int ntile = n >> 4, nl = n & 15;
  const int kstep = k >> 5, kk = k & 31;
  const int hi = (kk >> 3) & 1;
  const int e  = (kk & 7) | ((kk >> 4) << 3);
  return (((kstep * ntiles) + ntile) * 32 + (hi * 16 + nl)) * 16 + e;
}

__device__ __forceinline__ v8f wmma_bf16(v16bf a, v16bf b, v8f c) {
  // (neg_a, A, neg_b, B, c_mod, C, reuse_a, reuse_b)
  return __builtin_amdgcn_wmma_f32_16x16x32_bf16(false, a, false, b,
                                                 (short)0, c, false, false);
}

// ------------------------ degree / normalization ---------------------------
__global__ void k_deg_init(float* deg) {
  int n = blockIdx.x * blockDim.x + threadIdx.x;
  if (n < NN) deg[n] = 1.0f;  // self-loop contributes 1 to every node degree
}

__global__ void k_deg_accum(const long long* ei, float* deg, int E) {
  int e = blockIdx.x * blockDim.x + threadIdx.x;
  if (e < E) atomicAdd(&deg[(int)ei[E + e]], 1.0f);  // dst row
}

__global__ void k_dinv(const float* deg, float* dinv) {
  int n = blockIdx.x * blockDim.x + threadIdx.x;
  if (n < NN) {
    float d = deg[n];
    dinv[n] = (d > 0.0f) ? rsqrtf(d) : 0.0f;
  }
}

// --------------------------- temporal conv 1 -------------------------------
// y1[m, co, s] = b1[co] + sum_{ci,k} W1[co,ci,k] * x[b,ci,n,s+k-1]   (m=b*N+n)
// GEMM per node: [64 x 96] x [96 x 32]; 8 waves -> 8 16x16 tiles; 3 K-steps.
__global__ void k_tconv1(const float* __restrict__ x, const float* __restrict__ W1,
                         const float* __restrict__ b1, float* __restrict__ y1) {
  const int m = blockIdx.x, b = m >> 10, n = m & 1023;
  const int tid = threadIdx.x;
  __shared__ v16bf Afrag[4 * 3 * 32];   // W1 64x96, fragment-major
  __shared__ v16bf Bfrag[3 * 2 * 32];   // im2col 96x32, fragment-major
  __bf16* Af = (__bf16*)Afrag;
  __bf16* Bf = (__bf16*)Bfrag;

  if (tid == 0) __builtin_prefetch(W1, 0, 1);
  for (int i = tid; i < COUT * CIN * KW; i += 256) {
    const int co = i / (CIN * KW), k = i - co * (CIN * KW);
    Af[a_swz(co, k, 3)] = (__bf16)W1[i];
  }
  for (int i = tid; i < CIN * KW * SS; i += 256) {
    const int r = i >> 5, s = i & 31;
    const int ci = r / 3, k = r - ci * 3;
    const int ss = s + k - 1;
    float v = (ss >= 0 && ss < SS)
                  ? x[(((size_t)b * CIN + ci) * NN + n) * SS + ss] : 0.0f;
    Bf[b_swz(r, s, 2)] = (__bf16)v;
  }
  __syncthreads();

  const int w = tid >> 5, lane = tid & 31;
  const int mtile = w >> 1, ntile = w & 1;
  v8f acc = {};
#pragma unroll
  for (int ks = 0; ks < 3; ++ks)
    acc = wmma_bf16(Afrag[(mtile * 3 + ks) * 32 + lane],
                    Bfrag[(ks * 2 + ntile) * 32 + lane], acc);

  const int col = ntile * 16 + (lane & 15);
  const int hi  = (lane >> 4) & 1;
#pragma unroll
  for (int v = 0; v < 8; ++v) {
    const int co = mtile * 16 + hi * 8 + v;
    y1[((size_t)m * COUT + co) * SS + col] = acc[v] + b1[co];
  }
}

// -------------------- per-channel BN statistics (axes 0,2) -----------------
__global__ void k_stats_ch(const float* __restrict__ y, float* __restrict__ stat) {
  const int c = blockIdx.x, tid = threadIdx.x;
  const int total = BB * NN * SS;
  float s0 = 0.f, s1 = 0.f;
  for (int i = tid; i < total; i += 256) {
    const int mm = i >> 5, s = i & 31;
    float v = y[((size_t)mm * COUT + c) * SS + s];
    s0 += v; s1 += v * v;
  }
  __shared__ float r0[256], r1[256];
  r0[tid] = s0; r1[tid] = s1; __syncthreads();
  for (int off = 128; off > 0; off >>= 1) {
    if (tid < off) { r0[tid] += r0[tid + off]; r1[tid] += r1[tid + off]; }
    __syncthreads();
  }
  if (tid == 0) {
    float mean = r0[0] / (float)total;
    float var  = r1[0] / (float)total - mean * mean;
    stat[c * 2 + 0] = mean;
    stat[c * 2 + 1] = rsqrtf(var + EPSV);
  }
}

// -------------------- spatial projection: hw = relu(bn1(h)) @ Wg -----------
// A: 32 (nodes) x 64 (c) with BN+ReLU fused in the stage; B: Wg 64x64.
__global__ void k_hw(const float* __restrict__ y1, const float* __restrict__ stat1,
                     const float* __restrict__ g1, const float* __restrict__ be1,
                     const float* __restrict__ Wg, float* __restrict__ hw) {
  const int bs = blockIdx.x >> 5;      // b*32+s  (0..127)
  const int chunk = blockIdx.x & 31;   // node chunk of 32
  const int b = bs >> 5, s = bs & 31;
  const int tid = threadIdx.x;
  __shared__ v16bf Afrag[2 * 2 * 32];  // 32 x 64, fragment-major
  __shared__ v16bf Bfrag[2 * 4 * 32];  // Wg 64 x 64, fragment-major
  __bf16* Af = (__bf16*)Afrag;
  __bf16* Bf = (__bf16*)Bfrag;

  if (tid == 0) __builtin_prefetch(Wg, 0, 1);
  for (int i = tid; i < 32 * CSP; i += 256) {
    const int r = i >> 6, c = i & 63;
    const int n = chunk * 32 + r;
    float v = y1[(((size_t)b * NN + n) * COUT + c) * SS + s];
    v = (v - stat1[c * 2]) * stat1[c * 2 + 1] * g1[c] + be1[c];
    Af[a_swz(r, c, 2)] = (__bf16)fmaxf(v, 0.0f);
  }
  for (int i = tid; i < COUT * CSP; i += 256) {
    const int k = i >> 6, d = i & 63;
    Bf[b_swz(k, d, 4)] = (__bf16)Wg[i];
  }
  __syncthreads();

  const int w = tid >> 5, lane = tid & 31;
  const int mtile = w >> 2, ntile = w & 3;   // 2 x 4 tiles
  v8f acc = {};
#pragma unroll
  for (int ks = 0; ks < 2; ++ks)
    acc = wmma_bf16(Afrag[(mtile * 2 + ks) * 32 + lane],
                    Bfrag[(ks * 4 + ntile) * 32 + lane], acc);

  const int col = ntile * 16 + (lane & 15);
  const int hi  = (lane >> 4) & 1;
#pragma unroll
  for (int v = 0; v < 8; ++v) {
    const int n = chunk * 32 + mtile * 16 + hi * 8 + v;
    hw[((size_t)bs * NN + n) * CSP + col] = acc[v];
  }
}

// ----------------- agg init: bias + (non-atomic) self-loop term ------------
__global__ void k_agg_init(const float* __restrict__ hw, const float* __restrict__ dinv,
                           const float* __restrict__ bg, float* __restrict__ agg) {
  const size_t idx = (size_t)blockIdx.x * 256 + threadIdx.x;
  const int d = idx & 63;
  const int n = (idx >> 6) & 1023;
  const float di = dinv[n];
  agg[idx] = bg[d] + hw[idx] * di * di;
}

// ----------------- edge scatter: agg[dst] += hw[src] * norm ----------------
__global__ void k_msg(const long long* __restrict__ ei, const float* __restrict__ dinv,
                      const float* __restrict__ hw, float* __restrict__ agg, int E) {
  const int e = blockIdx.x;
  const int src = (int)ei[e];
  const int dst = (int)ei[E + e];
  const float norm = dinv[src] * dinv[dst];
  for (int i = threadIdx.x; i < BB * SS * CSP; i += 256) {
    const int bs = i >> 6, d = i & 63;
    atomicAdd(&agg[((size_t)bs * NN + dst) * CSP + d],
              hw[((size_t)bs * NN + src) * CSP + d] * norm);
  }
}

// ---------------- per-(b,s,d) BN statistics over the node dim --------------
__global__ void k_stats_bs(const float* __restrict__ agg, float* __restrict__ stat) {
  const int g = blockIdx.x;            // (b*32+s)*64 + d
  const int bs = g >> 6, d = g & 63;
  const float* p = agg + (size_t)bs * NN * CSP + d;
  const int tid = threadIdx.x;
  float s0 = 0.f, s1 = 0.f;
  for (int n = tid; n < NN; n += 256) {
    float v = p[(size_t)n * CSP];
    s0 += v; s1 += v * v;
  }
  __shared__ float r0[256], r1[256];
  r0[tid] = s0; r1[tid] = s1; __syncthreads();
  for (int off = 128; off > 0; off >>= 1) {
    if (tid < off) { r0[tid] += r0[tid + off]; r1[tid] += r1[tid + off]; }
    __syncthreads();
  }
  if (tid == 0) {
    float mean = r0[0] / (float)NN;
    float var  = r1[0] / (float)NN - mean * mean;
    stat[g * 2 + 0] = mean;
    stat[g * 2 + 1] = rsqrtf(var + EPSV);
  }
}

// --------------------------- temporal conv 2 -------------------------------
// BN_s + ReLU + im2col fused into the LDS stage; GEMM [64 x 192] x [192 x 32].
__global__ void k_tconv2(const float* __restrict__ agg, const float* __restrict__ stat_s,
                         const float* __restrict__ gs, const float* __restrict__ bes,
                         const float* __restrict__ W2, const float* __restrict__ b2,
                         float* __restrict__ y2) {
  const int m = blockIdx.x, b = m >> 10, n = m & 1023;
  const int tid = threadIdx.x;
  __shared__ v16bf Afrag[4 * 6 * 32];  // W2 64x192, fragment-major (24KB)
  __shared__ v16bf Bfrag[6 * 2 * 32];  // im2col 192x32, fragment-major (12KB)
  __bf16* Af = (__bf16*)Afrag;
  __bf16* Bf = (__bf16*)Bfrag;

  if (tid == 0) __builtin_prefetch(W2, 0, 1);
  for (int i = tid; i < COUT * CSP * KW; i += 256) {
    const int co = i / (CSP * KW), k = i - co * (CSP * KW);
    Af[a_swz(co, k, 6)] = (__bf16)W2[i];
  }
  for (int i = tid; i < CSP * KW * SS; i += 256) {
    const int r = i >> 5, s = i & 31;
    const int c = r / 3, k = r - c * 3;
    const int ss = s + k - 1;
    float v = 0.0f;
    if (ss >= 0 && ss < SS) {
      const int bs = b * SS + ss;
      float u = agg[((size_t)bs * NN + n) * CSP + c];
      const int g = bs * CSP + c;
      u = (u - stat_s[g * 2]) * stat_s[g * 2 + 1] * gs[c] + bes[c];
      v = fmaxf(u, 0.0f);
    }
    Bf[b_swz(r, s, 2)] = (__bf16)v;
  }
  __syncthreads();

  const int w = tid >> 5, lane = tid & 31;
  const int mtile = w >> 1, ntile = w & 1;
  v8f acc = {};
#pragma unroll
  for (int ks = 0; ks < 6; ++ks)
    acc = wmma_bf16(Afrag[(mtile * 6 + ks) * 32 + lane],
                    Bfrag[(ks * 2 + ntile) * 32 + lane], acc);

  const int col = ntile * 16 + (lane & 15);
  const int hi  = (lane >> 4) & 1;
#pragma unroll
  for (int v = 0; v < 8; ++v) {
    const int co = mtile * 16 + hi * 8 + v;
    y2[((size_t)m * COUT + co) * SS + col] = acc[v] + b2[co];
  }
}

// --------- final: residual 1x1 conv (WMMA) + bn2 + relu + add + relu -------
__global__ void k_final(const float* __restrict__ x, const float* __restrict__ Wres,
                        const float* __restrict__ bres, const float* __restrict__ y2,
                        const float* __restrict__ stat2, const float* __restrict__ g2,
                        const float* __restrict__ be2, float* __restrict__ out) {
  const int m = blockIdx.x, b = m >> 10, n = m & 1023;
  const int tid = threadIdx.x;
  __shared__ v16bf Afrag[4 * 1 * 32];  // Wres 64x32
  __shared__ v16bf Bfrag[1 * 2 * 32];  // x tile 32x32
  __bf16* Af = (__bf16*)Afrag;
  __bf16* Bf = (__bf16*)Bfrag;

  if (tid == 0) __builtin_prefetch(Wres, 0, 1);
  for (int i = tid; i < COUT * CIN; i += 256) {
    const int co = i >> 5, k = i & 31;
    Af[a_swz(co, k, 1)] = (__bf16)Wres[i];
  }
  for (int i = tid; i < CIN * SS; i += 256) {
    const int ci = i >> 5, s = i & 31;
    Bf[b_swz(ci, s, 2)] =
        (__bf16)x[(((size_t)b * CIN + ci) * NN + n) * SS + s];
  }
  __syncthreads();

  const int w = tid >> 5, lane = tid & 31;
  const int mtile = w >> 1, ntile = w & 1;
  v8f acc = {};
  acc = wmma_bf16(Afrag[mtile * 32 + lane], Bfrag[ntile * 32 + lane], acc);

  const int col = ntile * 16 + (lane & 15);
  const int hi  = (lane >> 4) & 1;
#pragma unroll
  for (int v = 0; v < 8; ++v) {
    const int co = mtile * 16 + hi * 8 + v;
    const float res = acc[v] + bres[co];
    float u = y2[((size_t)m * COUT + co) * SS + col];
    u = (u - stat2[co * 2]) * stat2[co * 2 + 1] * g2[co] + be2[co];
    u = fmaxf(u, 0.0f);                 // h2 = relu(bn2(conv2))
    out[(((size_t)b * COUT + co) * NN + n) * SS + col] = fmaxf(u + res, 0.0f);
  }
}

// ---------------------------------------------------------------------------
extern "C" void kernel_launch(void* const* d_in, const int* in_sizes, int n_in,
                              void* d_out, int out_size, void* d_ws, size_t ws_size,
                              hipStream_t stream) {
  const float*     x    = (const float*)d_in[0];
  const long long* ei   = (const long long*)d_in[1];
  const float*     W1   = (const float*)d_in[2];
  const float*     b1   = (const float*)d_in[3];
  const float*     g1   = (const float*)d_in[4];
  const float*     be1  = (const float*)d_in[5];
  const float*     Wg   = (const float*)d_in[6];
  const float*     bg   = (const float*)d_in[7];
  const float*     gs   = (const float*)d_in[8];
  const float*     bes  = (const float*)d_in[9];
  const float*     W2   = (const float*)d_in[10];
  const float*     b2   = (const float*)d_in[11];
  const float*     g2   = (const float*)d_in[12];
  const float*     be2  = (const float*)d_in[13];
  const float*     Wres = (const float*)d_in[14];
  const float*     bres = (const float*)d_in[15];
  float*           out  = (float*)d_out;
  const int E = in_sizes[1] / 2;

  // Workspace: two 32MB ping-pong tensors stay L2-resident (192 MB L2).
  const size_t BIG = (size_t)BB * SS * NN * CSP;  // 8,388,608
  float* ws    = (float*)d_ws;
  float* wsA   = ws;              // y1, then agg
  float* wsB   = ws + BIG;        // hw, then y2
  float* deg   = ws + 2 * BIG;
  float* dinv  = deg + NN;
  float* stat1 = dinv + NN;            // 64 x {mean, inv_std}
  float* stats = stat1 + 2 * COUT;     // 8192 x {mean, inv_std}
  float* stat2 = stats + 2 * BB * SS * CSP;

  const int NBLK = BB * NN;  // 4096 node-blocks

  // graph normalization
  k_deg_init<<<(NN + 255) / 256, 256, 0, stream>>>(deg);
  k_deg_accum<<<(E + 255) / 256, 256, 0, stream>>>(ei, deg, E);
  k_dinv<<<(NN + 255) / 256, 256, 0, stream>>>(deg, dinv);

  // temporal1 + BN stats
  k_tconv1<<<NBLK, 256, 0, stream>>>(x, W1, b1, wsA);
  k_stats_ch<<<COUT, 256, 0, stream>>>(wsA, stat1);

  // spatial projection (BN1+ReLU fused)
  k_hw<<<BB * SS * (NN / 32), 256, 0, stream>>>(wsA, stat1, g1, be1, Wg, wsB);

  // GCN aggregation: init(+bias+self-loop), edge scatter, per-(b,s,d) BN stats
  k_agg_init<<<(int)(BIG / 256), 256, 0, stream>>>(wsB, dinv, bg, wsA);
  k_msg<<<E, 256, 0, stream>>>(ei, dinv, wsB, wsA, E);
  k_stats_bs<<<BB * SS * CSP, 256, 0, stream>>>(wsA, stats);

  // temporal2 (BN_s+ReLU+im2col fused) + BN stats
  k_tconv2<<<NBLK, 256, 0, stream>>>(wsA, stats, gs, bes, W2, b2, wsB);
  k_stats_ch<<<COUT, 256, 0, stream>>>(wsB, stat2);

  // residual 1x1 conv (WMMA) + bn2 + relu + add + relu + layout transpose
  k_final<<<NBLK, 256, 0, stream>>>(x, Wres, bres, wsB, stat2, g2, be2, out);
}
